// Forecaster_48387101557242
// MI455X (gfx1250) — matrix operand order repaired and checked by reading, not compiled, not run on hardware
//
#include <hip/hip_runtime.h>
#include <hip/hip_bf16.h>

// ---------------------------------------------------------------------------
// LSTM forecaster for MI455X (gfx1250, wave32, WMMA).
// Persistent kernel: 64 workgroups x 256 threads; each WG owns a 16-row batch
// tile and runs all 512 encode steps + 128 future steps with h/c in LDS.
// W_ih/W_hh live in LDS as bf16 (256 KB of the 320 KB WGP LDS).
// Matrix math: v_wmma_f32_16x16x32_bf16, f32 accumulate.
// Per k-tile, all 8 B fragments are loaded into distinct registers before the
// 8 WMMAs so the scheduler can use partial s_wait_dscnt instead of draining
// LDS per matrix op; phase-1 inputs are double-buffered so global latency
// overlaps the matrix work.
// ---------------------------------------------------------------------------

#define HDIM    128
#define FOURH   512
#define BATCH   1024
#define TSTEPS  512
#define FUTURE  128
#define BM      16      // batch rows per workgroup

typedef __attribute__((ext_vector_type(16))) __bf16 v16bf;
typedef __attribute__((ext_vector_type(8)))  float  v8f;

// LDS layout (bytes)
#define OFF_WIH   0         // 512*128 bf16  = 131072
#define OFF_WHH   131072    // 512*128 bf16  = 131072
#define OFF_G     262144    // 16*512  f32   = 32768
#define OFF_C     294912    // 16*128  f32   = 8192
#define OFF_H     303104    // 16*128  bf16  = 4096
#define OFF_X0    307200    // 16*128  bf16  = 4096  (ping)
#define OFF_X1    311296    // 16*128  bf16  = 4096  (pong)
#define OFF_B     315392    // 512     f32   = 2048
#define SMEM_BYTES 317440   // <= 320 KB WGP LDS

union FragBF { v16bf v; uint4 q[2]; };

__device__ __forceinline__ float fast_tanh(float x) {
    x = fminf(fmaxf(x, -15.0f), 15.0f);
    const float e = __expf(2.0f * x);
    return (e - 1.0f) / (e + 1.0f);
}

__device__ __forceinline__ float fast_sigmoid(float x) {
    return 1.0f / (1.0f + __expf(-x));
}

__device__ __forceinline__ v16bf load_a_frag(const __bf16* base, int m, int half, int kt) {
    // A-matrix 16x32 bf16: lane m, lane-half 'half'; 16 contiguous K at half*16
    FragBF f;
    const int k0 = kt * 32 + half * 16;
    f.q[0] = *(const uint4*)(base + m * HDIM + k0);
    f.q[1] = *(const uint4*)(base + m * HDIM + k0 + 8);
    return f.v;
}

__device__ __forceinline__ v16bf load_b_frag(const __bf16* wRowMajorNK, int ncol, int half, int kt) {
    // B-matrix 32x16: lane's column = ncol, 16 contiguous K at half*16
    FragBF f;
    const int kb = kt * 32 + half * 16;
    f.q[0] = *(const uint4*)(wRowMajorNK + ncol * HDIM + kb);
    f.q[1] = *(const uint4*)(wRowMajorNK + ncol * HDIM + kb + 8);
    return f.v;
}

// Pack 8 f32 -> 8 bf16 and store as one 16-byte LDS write.
__device__ __forceinline__ void store8_bf16(__bf16* dst, float4 a, float4 b) {
    union { uint4 q; __bf16 h[8]; } u;
    u.h[0] = (__bf16)a.x; u.h[1] = (__bf16)a.y; u.h[2] = (__bf16)a.z; u.h[3] = (__bf16)a.w;
    u.h[4] = (__bf16)b.x; u.h[5] = (__bf16)b.y; u.h[6] = (__bf16)b.z; u.h[7] = (__bf16)b.w;
    *(uint4*)dst = u.q;
}

// One LSTM step for this WG's 16-row tile: gates = x@Wih^T + h@Whh^T + bias,
// then the elementwise cell update. All operands LDS-resident.
__device__ __forceinline__ void lstm_step(const __bf16* sWih, const __bf16* sWhh,
                                          float* sG, float* sC, __bf16* sH,
                                          const __bf16* sX, const float* sB, int tid) {
    const int wave = tid >> 5;
    const int lane = tid & 31;
    const int m    = lane & 15;
    const int half = lane >> 4;

    // ---- gates GEMM: each wave owns 4 N-tiles of the [16 x 512] output ----
    v8f acc[4];
#pragma unroll
    for (int j = 0; j < 4; ++j) {
        const float bias = sB[(wave * 4 + j) * 16 + m];  // column-only bias
#pragma unroll
        for (int r = 0; r < 8; ++r) acc[j][r] = bias;
    }

#pragma unroll
    for (int kt = 0; kt < 4; ++kt) {
        // Issue ALL fragment loads for this k-tile first (distinct registers)
        // so the 16+4 ds_load_b128 pipeline and the WMMAs below only take
        // partial s_wait_dscnt waits.
        v16bf ax = load_a_frag(sX, m, half, kt);
        v16bf ah = load_a_frag(sH, m, half, kt);
        v16bf bhh[4], bih[4];
#pragma unroll
        for (int j = 0; j < 4; ++j) {
            const int ncol = (wave * 4 + j) * 16 + m;
            bhh[j] = load_b_frag(sWhh, ncol, half, kt);
            bih[j] = load_b_frag(sWih, ncol, half, kt);
        }
#pragma unroll
        for (int j = 0; j < 4; ++j) {
            acc[j] = __builtin_amdgcn_wmma_f32_16x16x32_bf16(
                false, ah, false, bhh[j], (short)0, acc[j], false, false);
            acc[j] = __builtin_amdgcn_wmma_f32_16x16x32_bf16(
                false, ax, false, bih[j], (short)0, acc[j], false, false);
        }
    }

    // scatter gate fragments to LDS: elem r -> row (r + 8*half), col ncol
#pragma unroll
    for (int j = 0; j < 4; ++j) {
        const int ncol = (wave * 4 + j) * 16 + m;
#pragma unroll
        for (int r = 0; r < 8; ++r)
            sG[(r + half * 8) * FOURH + ncol] = acc[j][r];
    }
    __syncthreads();

    // ---- elementwise cell update (i,f,g,o order), 8 elems per thread ----
    for (int e = tid; e < BM * HDIM; e += 256) {
        const int mm = e >> 7;
        const int jj = e & (HDIM - 1);
        const float gi = sG[mm * FOURH + jj];
        const float gf = sG[mm * FOURH + HDIM + jj];
        const float gg = sG[mm * FOURH + 2 * HDIM + jj];
        const float go = sG[mm * FOURH + 3 * HDIM + jj];
        const float i_ = fast_sigmoid(gi);
        const float f_ = fast_sigmoid(gf);
        const float o_ = fast_sigmoid(go);
        const float g_ = fast_tanh(gg);
        const float c_ = f_ * sC[e] + i_ * g_;
        sC[e] = c_;
        sH[e] = (__bf16)(o_ * fast_tanh(c_));
    }
    __syncthreads();
}

// Projection out[16,128] = h @ W_lin^T + b_lin; one 16x16 N-tile per wave.
// Writes f32 to global output row and bf16 back into sX (next step's input).
__device__ __forceinline__ void project(const __bf16* sH, __bf16* sX,
                                        const float* __restrict__ W_lin,
                                        const float* __restrict__ b_lin,
                                        float* __restrict__ outRow, int tid) {
    const int wave = tid >> 5;
    const int lane = tid & 31;
    const int m    = lane & 15;
    const int half = lane >> 4;
    const int ncol = wave * 16 + m;

    v8f acc;
    const float bias = b_lin[ncol];
#pragma unroll
    for (int r = 0; r < 8; ++r) acc[r] = bias;

#pragma unroll
    for (int kt = 0; kt < 4; ++kt) {
        v16bf ah = load_a_frag(sH, m, half, kt);
        // B fragment: 16 contiguous f32 of W_lin row 'ncol' (L2-resident),
        // converted to bf16 on the fly
        const float4* w4 = (const float4*)(W_lin + ncol * HDIM + kt * 32 + half * 16);
        float4 w0 = w4[0], w1 = w4[1], w2 = w4[2], w3 = w4[3];
        v16bf bv;
        bv[0]  = (__bf16)w0.x; bv[1]  = (__bf16)w0.y; bv[2]  = (__bf16)w0.z; bv[3]  = (__bf16)w0.w;
        bv[4]  = (__bf16)w1.x; bv[5]  = (__bf16)w1.y; bv[6]  = (__bf16)w1.z; bv[7]  = (__bf16)w1.w;
        bv[8]  = (__bf16)w2.x; bv[9]  = (__bf16)w2.y; bv[10] = (__bf16)w2.z; bv[11] = (__bf16)w2.w;
        bv[12] = (__bf16)w3.x; bv[13] = (__bf16)w3.y; bv[14] = (__bf16)w3.z; bv[15] = (__bf16)w3.w;
        acc = __builtin_amdgcn_wmma_f32_16x16x32_bf16(
            false, ah, false, bv, (short)0, acc, false, false);
    }

#pragma unroll
    for (int r = 0; r < 8; ++r) {
        const int row = r + half * 8;
        outRow[row * HDIM + ncol] = acc[r];
        sX[row * HDIM + ncol]     = (__bf16)acc[r];
    }
    __syncthreads();
}

__global__ __launch_bounds__(256)
void lstm_forecast_kernel(const float* __restrict__ inp,
                          const float* __restrict__ W_ih,
                          const float* __restrict__ W_hh,
                          const float* __restrict__ b_ih,
                          const float* __restrict__ b_hh,
                          const float* __restrict__ W_lin,
                          const float* __restrict__ b_lin,
                          float* __restrict__ out) {
    extern __shared__ char smem[];
    __bf16* sWih = (__bf16*)(smem + OFF_WIH);
    __bf16* sWhh = (__bf16*)(smem + OFF_WHH);
    float*  sG   = (float*)(smem + OFF_G);
    float*  sC   = (float*)(smem + OFF_C);
    __bf16* sH   = (__bf16*)(smem + OFF_H);
    __bf16* sX0  = (__bf16*)(smem + OFF_X0);
    __bf16* sX1  = (__bf16*)(smem + OFF_X1);
    float*  sB   = (float*)(smem + OFF_B);

    const int tid    = threadIdx.x;
    const int batch0 = blockIdx.x * BM;   // this WG owns batch rows [batch0, batch0+16)

    // One-time: weights -> LDS bf16 (L2-resident f32 source), bias sum, h/c=0
    for (int i = tid; i < FOURH * HDIM; i += 256) {
        sWih[i] = (__bf16)W_ih[i];
        sWhh[i] = (__bf16)W_hh[i];
    }
    for (int i = tid; i < FOURH; i += 256) sB[i] = b_ih[i] + b_hh[i];
    for (int i = tid; i < BM * HDIM; i += 256) {
        sH[i] = (__bf16)0.0f;
        sC[i] = 0.0f;
    }

    // Prologue: stage x(t=0). Each thread owns 8 contiguous f32 of the tile.
    {
        const float4* p = (const float4*)(inp + (size_t)batch0 * HDIM) + tid * 2;
        store8_bf16(sX0 + tid * 8, p[0], p[1]);
    }
    __syncthreads();

    // ---- Phase 1: consume the 512 given inputs (double-buffered x) ----
    for (int t = 0; t < TSTEPS; ++t) {
        const __bf16* cur = (t & 1) ? sX1 : sX0;
        __bf16*       nxt = (t & 1) ? sX0 : sX1;

        // Issue next step's global loads now; LOADcnt lets them complete
        // while this step's 32 WMMAs run.
        float4 a = {}, b = {};
        const bool more = (t + 1 < TSTEPS);
        if (more) {
            const float4* p = (const float4*)(inp + ((size_t)(t + 1) * BATCH + batch0) * HDIM) + tid * 2;
            a = p[0];
            b = p[1];
        }

        lstm_step(sWih, sWhh, sG, sC, sH, cur, sB, tid);

        // Commit next input into the inactive buffer; visible after the
        // __syncthreads() below.
        if (more) store8_bf16(nxt + tid * 8, a, b);
        __syncthreads();
    }

    // out[0] = h @ W_lin^T + b_lin (also seeds sX0 for the future phase)
    project(sH, sX0, W_lin, b_lin, out + (size_t)batch0 * HDIM, tid);

    // ---- Phase 2: 128 autoregressive future steps ----
    for (int f = 0; f < FUTURE; ++f) {
        lstm_step(sWih, sWhh, sG, sC, sH, sX0, sB, tid);
        project(sH, sX0, W_lin, b_lin,
                out + ((size_t)(1 + f) * BATCH + batch0) * HDIM, tid);
    }
}

extern "C" void kernel_launch(void* const* d_in, const int* in_sizes, int n_in,
                              void* d_out, int out_size, void* d_ws, size_t ws_size,
                              hipStream_t stream) {
    (void)in_sizes; (void)n_in; (void)d_ws; (void)ws_size; (void)out_size;
    const float* inp   = (const float*)d_in[0];
    const float* W_ih  = (const float*)d_in[1];
    const float* W_hh  = (const float*)d_in[2];
    const float* b_ih  = (const float*)d_in[3];
    const float* b_hh  = (const float*)d_in[4];
    const float* W_lin = (const float*)d_in[5];
    const float* b_lin = (const float*)d_in[6];
    float* out = (float*)d_out;

    // Allow >default dynamic LDS (310 KB of the 320 KB per-WGP pool).
    hipFuncSetAttribute((const void*)lstm_forecast_kernel,
                        hipFuncAttributeMaxDynamicSharedMemorySize, SMEM_BYTES);

    dim3 grid(BATCH / BM);   // 64 persistent workgroups, one per WGP
    dim3 block(256);         // 8 wave32s
    lstm_forecast_kernel<<<grid, block, SMEM_BYTES, stream>>>(
        inp, W_ih, W_hh, b_ih, b_hh, W_lin, b_lin, out);
}
